// CausalSelfAttention_41652592836854
// MI455X (gfx1250) — matrix operand order repaired and checked by reading, not compiled, not run on hardware
//
#include <hip/hip_runtime.h>

// ---------------------------------------------------------------------------
// CDNA5 / gfx1250 causal self-attention forward.
// Sizes (hardcoded from reference): B=2, T=2048, D=2048, H=16, HD=128.
// bf16 WMMA (v_wmma_f32_16x16x32_bf16) for QKV GEMM, flash attention
// (QK^T and PV), and output projection; f32 accumulation everywhere.
// Round 3: fix async builtin arg types (toolchain wants generic v4i*).
// ---------------------------------------------------------------------------

typedef __attribute__((ext_vector_type(16))) __bf16 v16bf;
typedef __attribute__((ext_vector_type(8)))  float  v8f;

#define B_    2
#define T_    2048
#define D_    2048
#define H_    16
#define HD_   128
#define HALF_ 64

// ---- async global->LDS staging (CDNA5 ASYNCcnt path), with fallback -------
#if defined(__has_builtin)
#if __has_builtin(__builtin_amdgcn_global_load_async_to_lds_b128)
#define HAVE_ASYNC_LDS 1
#endif
#endif

#ifdef HAVE_ASYNC_LDS
typedef int v4i_ __attribute__((vector_size(4 * sizeof(int))));
#define ASYNC_COPY16(dst_lds, src_glb)                                        \
    __builtin_amdgcn_global_load_async_to_lds_b128(                           \
        (v4i_*)(src_glb), (v4i_*)(dst_lds), 0, 0)
#if __has_builtin(__builtin_amdgcn_s_wait_asynccnt)
#define ASYNC_WAIT() __builtin_amdgcn_s_wait_asynccnt(0)
#else
#define ASYNC_WAIT() asm volatile("s_wait_asynccnt 0x0" ::: "memory")
#endif
#else
#define ASYNC_COPY16(dst_lds, src_glb)                                        \
    (*(uint4*)(dst_lds) = *(const uint4*)(src_glb))
#define ASYNC_WAIT() ((void)0)
#endif

__device__ static inline __bf16 f2bf(float f) { return (__bf16)f; }

__device__ static inline v8f wmma_bf16(v16bf a, v16bf b, v8f c) {
    return __builtin_amdgcn_wmma_f32_16x16x32_bf16(
        /*neg_a=*/false, a, /*neg_b=*/false, b,
        /*c_mod=*/(short)0, c, /*reuse_a=*/false, /*reuse_b=*/false);
}

// A fragment (16x32 bf16, MxK). Layout: lanes 0-15 -> M=lane, K=0..15;
// lanes 16-31 -> same M, K=16..31 (matches B fragment); each VGPR
// packs 2 consecutive K. LDS tile is row-major [m][k]; both 16-byte groups
// per lane are 16B-aligned so they coalesce into ds_load_b128.
__device__ static inline v16bf load_a_frag(const __bf16* base, int ld,
                                           int row0, int k0, int lane) {
    const int m    = lane & 15;
    const int half = lane >> 4;
    const __bf16* rp = base + (size_t)(row0 + m) * ld + k0;
    const int kb0 = half * 16;
    const int kb1 = half * 16 + 8;
    union { v16bf v; unsigned u[8]; } f;
#pragma unroll
    for (int i = 0; i < 4; ++i) {
        f.u[i]     = *(const unsigned*)(rp + kb0 + 2 * i);
        f.u[4 + i] = *(const unsigned*)(rp + kb1 + 2 * i);
    }
    return f.v;
}

// B fragment (32x16 bf16, KxN). Lanes 0-15: K=0..15, lanes 16-31: K=16..31,
// N = lane&15, 16 contiguous K per lane. LDS tile stored transposed [n][k].
__device__ static inline v16bf load_b_frag(const __bf16* base, int ld,
                                           int col0, int k0, int lane) {
    const int n  = lane & 15;
    const int kg = (lane >> 4) * 16;
    const __bf16* p = base + (size_t)(col0 + n) * ld + k0 + kg;
    union { v16bf v; unsigned u[8]; } f;
#pragma unroll
    for (int i = 0; i < 8; ++i) f.u[i] = *(const unsigned*)(p + 2 * i);
    return f.v;
}

// ---------------------------------------------------------------------------
// GEMM: C(f32, MxN) = A(f32, MxK) * B(f32, KxN) with bf16 WMMA.
// Block tile 128x256, K-step 32, 256 threads = 8 waves in a 2x4 grid; each
// wave owns a 64x64 sub-tile (16 accumulators). Double-buffered LDS: next
// tile's f32 global loads are issued into registers before the WMMA burst,
// converted+stored to the other buffer after it. One barrier per K-step.
// ---------------------------------------------------------------------------
#define G_TM 128
#define G_TN 256
#define G_TK 32
#define G_LD (G_TK + 8)   // 80B row stride (16B aligned)

__device__ static inline void gemm_glb_load(
    const float* __restrict__ A, const float* __restrict__ Bm,
    int N, int K, int m0, int n0, int k0, int tid,
    float4* a_reg, float4* b_reg) {
#pragma unroll
    for (int j = 0; j < 4; ++j) {             // A: 128x32 = 1024 float4
        const int ci = tid + j * 256;
        const int r = ci >> 3, c = (ci & 7) * 4;
        a_reg[j] = *(const float4*)(A + (size_t)(m0 + r) * K + k0 + c);
    }
#pragma unroll
    for (int j = 0; j < 8; ++j) {             // B: 32x256 = 2048 float4
        const int ci = tid + j * 256;
        const int kr = ci >> 6, nc = (ci & 63) * 4;
        b_reg[j] = *(const float4*)(Bm + (size_t)(k0 + kr) * N + n0 + nc);
    }
}

__device__ static inline void gemm_lds_store(
    __bf16 (*As)[G_LD], __bf16 (*Bs)[G_LD], int tid,
    const float4* a_reg, const float4* b_reg) {
#pragma unroll
    for (int j = 0; j < 4; ++j) {
        const int ci = tid + j * 256;
        const int r = ci >> 3, c = (ci & 7) * 4;
        As[r][c + 0] = f2bf(a_reg[j].x); As[r][c + 1] = f2bf(a_reg[j].y);
        As[r][c + 2] = f2bf(a_reg[j].z); As[r][c + 3] = f2bf(a_reg[j].w);
    }
#pragma unroll
    for (int j = 0; j < 8; ++j) {
        const int ci = tid + j * 256;
        const int kr = ci >> 6, nc = (ci & 63) * 4;
        Bs[nc + 0][kr] = f2bf(b_reg[j].x); Bs[nc + 1][kr] = f2bf(b_reg[j].y);
        Bs[nc + 2][kr] = f2bf(b_reg[j].z); Bs[nc + 3][kr] = f2bf(b_reg[j].w);
    }
}

__global__ __launch_bounds__(256) void gemm_bf16_wmma(
    const float* __restrict__ A, const float* __restrict__ Bm,
    float* __restrict__ C, int M, int N, int K) {
    __shared__ __bf16 As[2][G_TM][G_LD];
    __shared__ __bf16 Bs[2][G_TN][G_LD];

    const int tid  = threadIdx.x;
    const int lane = tid & 31;
    const int w    = tid >> 5;
    const int wr   = w >> 2;     // 0..1 -> 64 rows
    const int wc   = w & 3;      // 0..3 -> 64 cols
    const int m0   = blockIdx.y * G_TM;
    const int n0   = blockIdx.x * G_TN;

    v8f acc[4][4];
#pragma unroll
    for (int i = 0; i < 4; ++i)
#pragma unroll
        for (int j = 0; j < 4; ++j)
#pragma unroll
            for (int e = 0; e < 8; ++e) acc[i][j][e] = 0.0f;

    float4 a_reg[4], b_reg[8];
    gemm_glb_load(A, Bm, N, K, m0, n0, 0, tid, a_reg, b_reg);
    gemm_lds_store(As[0], Bs[0], tid, a_reg, b_reg);
    __syncthreads();

    const int nk = K / G_TK;
    for (int kt = 0; kt < nk; ++kt) {
        const int cur = kt & 1;
        const bool has_next = (kt + 1) < nk;
        if (has_next)
            gemm_glb_load(A, Bm, N, K, m0, n0, (kt + 1) * G_TK, tid,
                          a_reg, b_reg);

        v16bf bf[4];
#pragma unroll
        for (int nt = 0; nt < 4; ++nt)
            bf[nt] = load_b_frag(&Bs[cur][0][0], G_LD, wc * 64 + nt * 16, 0,
                                 lane);
#pragma unroll
        for (int mt = 0; mt < 4; ++mt) {
            v16bf af = load_a_frag(&As[cur][0][0], G_LD, wr * 64 + mt * 16, 0,
                                   lane);
#pragma unroll
            for (int nt = 0; nt < 4; ++nt)
                acc[mt][nt] = wmma_bf16(af, bf[nt], acc[mt][nt]);
        }

        if (has_next)
            gemm_lds_store(As[cur ^ 1], Bs[cur ^ 1], tid, a_reg, b_reg);
        __syncthreads();
    }

    // C layout: VGPR r -> row r + 8*(lane>=16), col = lane&15. Pure int32
    // index math (max idx ~25M) -> 32-bit mads instead of v_mul_u64 chains.
    const int n_l  = lane & 15;
    const int mo   = (lane >> 4) * 8;
    const int base = (m0 + wr * 64 + mo) * N + n0 + wc * 64 + n_l;
#pragma unroll
    for (int mt = 0; mt < 4; ++mt)
#pragma unroll
        for (int r = 0; r < 8; ++r) {
            const int off = base + (mt * 16 + r) * N;
#pragma unroll
            for (int nt = 0; nt < 4; ++nt)
                C[(size_t)(off + nt * 16)] = acc[mt][nt][r];
        }
}

// ---------------------------------------------------------------------------
// Rotary + split: qkv (B,T,3,H,HD) f32 -> q,k (rotated) and v, bf16 (B,H,T,HD)
// ---------------------------------------------------------------------------
__global__ __launch_bounds__(256) void rotary_split(
    const float* __restrict__ qkv, const float* __restrict__ cosv,
    const float* __restrict__ sinv, __bf16* __restrict__ q,
    __bf16* __restrict__ k, __bf16* __restrict__ v) {
    const int idx = blockIdx.x * blockDim.x + threadIdx.x;
    const int i = idx & (HALF_ - 1);
    const int t = (idx >> 6) & (T_ - 1);
    const int h = (idx >> 17) & (H_ - 1);
    const int b = idx >> 21;

    const size_t base = ((size_t)(b * T_ + t) * 3) * D_;
    const float* qp = qkv + base + 0 * D_ + h * HD_;
    const float* kp = qkv + base + 1 * D_ + h * HD_;
    const float* vp = qkv + base + 2 * D_ + h * HD_;

    const float c = cosv[t * HALF_ + i];
    const float s = sinv[t * HALF_ + i];
    const float q1 = qp[i], q2 = qp[HALF_ + i];
    const float k1 = kp[i], k2 = kp[HALF_ + i];

    const size_t ob = (((size_t)b * H_ + h) * T_ + t) * HD_;
    q[ob + 2 * i]     = f2bf(q1 * c - q2 * s);
    q[ob + 2 * i + 1] = f2bf(q1 * s + q2 * c);
    k[ob + 2 * i]     = f2bf(k1 * c - k2 * s);
    k[ob + 2 * i + 1] = f2bf(k1 * s + k2 * c);
    v[ob + i]         = f2bf(vp[i]);
    v[ob + HALF_ + i] = f2bf(vp[HALF_ + i]);
}

// ---------------------------------------------------------------------------
// Flash attention: one block per (b*H+h, 128-row q tile). 256 threads = 8
// waves; wave w owns q rows [w*16, w*16+16). Online softmax, WMMA for both
// QK^T and PV. Q/K tiles staged with GLOBAL_LOAD_ASYNC_TO_LDS_B128 (pure
// byte copies); V needs an element transpose so it uses the normal path.
// P staging LDS is aliased over the K tile (dead after S is computed).
// ---------------------------------------------------------------------------
__global__ __launch_bounds__(256) void flash_attn(
    const __bf16* __restrict__ Qg, const __bf16* __restrict__ Kg,
    const __bf16* __restrict__ Vg, float* __restrict__ O) {
    constexpr int BQ = 128, BK = 128, LD = HD_ + 8;   // 272B rows, 16B aligned
    __shared__ __bf16 Qs[BQ][LD];      // [qrow][hd]
    __shared__ __bf16 Ks[BK][LD];      // [key][hd]; reused as P [qrow][key]
    __shared__ __bf16 Vs[HD_][BK + 8]; // [hd][key] (transposed for B-frag)

    const int tid  = threadIdx.x;
    const int lane = tid & 31;
    const int w    = tid >> 5;
    const int half = lane >> 4;
    const int n_l  = lane & 15;
    const int qt   = blockIdx.x;
    const int bh   = blockIdx.y;
    const int b    = bh >> 4;
    const int h    = bh & (H_ - 1);
    const size_t plane = (size_t)bh * (size_t)T_ * HD_;
    const int q0 = qt * BQ;
    const float scale = 0.08838834764831845f;   // 1/sqrt(128)

    // Load the Q tile once (async: 16B per issue, ASYNCcnt-tracked).
    for (int i = tid; i < BQ * HD_ / 8; i += 256) {
        const int r = i >> 4, c = (i & 15) * 8;
        ASYNC_COPY16(&Qs[r][c], Qg + plane + (size_t)(q0 + r) * HD_ + c);
    }

    v8f oacc[8];
#pragma unroll
    for (int nt = 0; nt < 8; ++nt)
#pragma unroll
        for (int e = 0; e < 8; ++e) oacc[nt][e] = 0.0f;
    float rmax[8], rsum[8];
#pragma unroll
    for (int r = 0; r < 8; ++r) { rmax[r] = -3.0e38f; rsum[r] = 0.0f; }

    __bf16* Pw = &Ks[w * 16][0];   // per-wave P staging (aliases K tile)

    for (int kt = 0; kt <= qt; ++kt) {
        const int k0 = kt * BK;
        __syncthreads();   // prior iteration done with Ks(P) and Vs
        for (int i = tid; i < BK * HD_ / 8; i += 256) {
            const int r = i >> 4, c = (i & 15) * 8;
            ASYNC_COPY16(&Ks[r][c], Kg + plane + (size_t)(k0 + r) * HD_ + c);
        }
        for (int i = tid; i < BK * HD_ / 8; i += 256) {
            const int r = i >> 4, c = (i & 15) * 8;
            union { uint4 q; __bf16 e[8]; } val;
            val.q = *(const uint4*)(Vg + plane + (size_t)(k0 + r) * HD_ + c);
#pragma unroll
            for (int j = 0; j < 8; ++j) Vs[c + j][r] = val.e[j];
        }
        ASYNC_WAIT();
        __syncthreads();

        // S = Q * K^T  (wave's 16 rows x 128 keys), HD split into 4 K-steps.
        v8f sacc[8];
#pragma unroll
        for (int nt = 0; nt < 8; ++nt)
#pragma unroll
            for (int e = 0; e < 8; ++e) sacc[nt][e] = 0.0f;
#pragma unroll
        for (int kk = 0; kk < 4; ++kk) {
            v16bf aq = load_a_frag(&Qs[0][0], LD, w * 16, kk * 32, lane);
#pragma unroll
            for (int nt = 0; nt < 8; ++nt) {
                v16bf bk = load_b_frag(&Ks[0][0], LD, nt * 16, kk * 32, lane);
                sacc[nt] = wmma_bf16(aq, bk, sacc[nt]);
            }
        }
        __syncthreads();   // all waves done reading Ks before P overwrites it

        // Scale + causal mask (only the diagonal k-tile needs masking).
#pragma unroll
        for (int nt = 0; nt < 8; ++nt)
#pragma unroll
            for (int r = 0; r < 8; ++r) {
                float sv = sacc[nt][r] * scale;
                if (kt == qt) {
                    const int qi = w * 16 + half * 8 + r;
                    const int ki = nt * 16 + n_l;
                    if (ki > qi) sv = -3.0e38f;
                }
                sacc[nt][r] = sv;
            }

        // Online softmax: row reductions stay in 16-lane halves (C layout).
#pragma unroll
        for (int r = 0; r < 8; ++r) {
            float m = rmax[r];
#pragma unroll
            for (int nt = 0; nt < 8; ++nt) m = fmaxf(m, sacc[nt][r]);
            for (int off = 8; off >= 1; off >>= 1)
                m = fmaxf(m, __shfl_xor(m, off, 32));
            const float f = __expf(rmax[r] - m);
            float ps = 0.0f;
#pragma unroll
            for (int nt = 0; nt < 8; ++nt) {
                const float p = __expf(sacc[nt][r] - m);
                sacc[nt][r] = p;
                ps += p;
            }
            for (int off = 8; off >= 1; off >>= 1)
                ps += __shfl_xor(ps, off, 32);
            rsum[r] = rsum[r] * f + ps;
            rmax[r] = m;
#pragma unroll
            for (int nt = 0; nt < 8; ++nt) oacc[nt][r] *= f;
        }

        // Stage P (16x128 bf16) into the wave's private LDS slice.
#pragma unroll
        for (int nt = 0; nt < 8; ++nt)
#pragma unroll
            for (int r = 0; r < 8; ++r)
                Pw[(half * 8 + r) * LD + nt * 16 + n_l] = f2bf(sacc[nt][r]);

        // O += P * V (keys are the contraction dim, 4 K-steps of 32).
#pragma unroll
        for (int kk = 0; kk < 4; ++kk) {
            v16bf ap = load_a_frag(Pw, LD, 0, kk * 32, lane);
#pragma unroll
            for (int nt = 0; nt < 8; ++nt) {
                v16bf bv = load_b_frag(&Vs[0][0], BK + 8, nt * 16, kk * 32,
                                       lane);
                oacc[nt] = wmma_bf16(ap, bv, oacc[nt]);
            }
        }
    }

    // Normalize and scatter to (B,T,D) f32 for the projection GEMM.
#pragma unroll
    for (int r = 0; r < 8; ++r) {
        const float inv = 1.0f / rsum[r];
        const int t = q0 + w * 16 + half * 8 + r;
        float* orow = O + ((size_t)b * T_ + t) * D_ + h * HD_;
#pragma unroll
        for (int nt = 0; nt < 8; ++nt)
            orow[nt * 16 + n_l] = oacc[nt][r] * inv;
    }
}

// ---------------------------------------------------------------------------
// Host-side launch.
// ---------------------------------------------------------------------------
extern "C" void kernel_launch(void* const* d_in, const int* in_sizes, int n_in,
                              void* d_out, int out_size, void* d_ws, size_t ws_size,
                              hipStream_t stream) {
    (void)in_sizes; (void)n_in; (void)out_size; (void)ws_size;
    const float* x     = (const float*)d_in[0];   // (B,T,D)
    const float* cosv  = (const float*)d_in[1];   // (T, HD/2)
    const float* sinv  = (const float*)d_in[2];   // (T, HD/2)
    const float* Wqkv  = (const float*)d_in[3];   // (D, 3D)
    const float* Wproj = (const float*)d_in[4];   // (D, D)
    float* out = (float*)d_out;                    // (B,T,D)

    char* ws = (char*)d_ws;
    const size_t qkv_bytes  = (size_t)B_ * T_ * 3 * D_ * 4;   // 100,663,296
    const size_t head_bytes = (size_t)B_ * H_ * T_ * HD_ * 2; //  16,777,216
    float*  qkv  = (float*)ws;
    __bf16* qb   = (__bf16*)(ws + qkv_bytes);
    __bf16* kb   = (__bf16*)(ws + qkv_bytes + head_bytes);
    __bf16* vb   = (__bf16*)(ws + qkv_bytes + 2 * head_bytes);
    float*  attn = (float*)(ws + qkv_bytes + 3 * head_bytes); // (B,T,D) f32

    // 1) QKV GEMM: (4096 x 2048) @ (2048 x 6144)
    gemm_bf16_wmma<<<dim3(3 * D_ / G_TN, B_ * T_ / G_TM), 256, 0, stream>>>(
        x, Wqkv, qkv, B_ * T_, 3 * D_, D_);

    // 2) Rotary + split to bf16 head-major layout.
    const int rot_threads = B_ * H_ * T_ * HALF_;
    rotary_split<<<rot_threads / 256, 256, 0, stream>>>(
        qkv, cosv, sinv, qb, kb, vb);

    // 3) Flash attention: grid = (q tiles, b*h)
    flash_attn<<<dim3(T_ / 128, B_ * H_), 256, 0, stream>>>(qb, kb, vb, attn);

    // 4) Output projection: (4096 x 2048) @ (2048 x 2048)
    gemm_bf16_wmma<<<dim3(D_ / G_TN, B_ * T_ / G_TM), 256, 0, stream>>>(
        attn, Wproj, out, B_ * T_, D_, D_);
}
